// Attention_3281355014171
// MI455X (gfx1250) — matrix operand order repaired and checked by reading, not compiled, not run on hardware
//
#include <hip/hip_runtime.h>
#include <hip/hip_bf16.h>

// ---------------------------------------------------------------------------
// Self-attention over B=16, C=512, H=W=32 (N=1024 tokens), fp32 in/out.
// All GEMMs via v_wmma_f32_16x16x32_f16 (CDNA5 / gfx1250, wave32).
// Q-tile staging via global_load_async_to_lds_b128 (ASYNCcnt path).
// ---------------------------------------------------------------------------

#define BATCH 16
#define CDIM  512
#define NTOK  1024
#define LROW  520    // padded row stride (halfs) for 512-wide LDS tiles
#define PROW  1032   // padded row stride (halfs) for 1024-wide LDS tiles

typedef __attribute__((ext_vector_type(16))) _Float16     v16h;
typedef __attribute__((ext_vector_type(8)))  float        v8f;
typedef __attribute__((ext_vector_type(8)))  unsigned int v8u;

#define WMMA_F16(ACC, A, Bm) \
    (ACC) = __builtin_amdgcn_wmma_f32_16x16x32_f16(false, (A), false, (Bm), (short)0, (ACC), false, false)

// A-matrix fragment (16x32 f16). Lane L holds row m=L&15;
// half = L>>4.  Elements 0..15 -> K = half*16 + i (contiguous, matches B).
// => two contiguous 8-half (16B) runs at base and base+8.
__device__ __forceinline__ v16h load_a16(const _Float16* base) {
    const uint4* p0 = (const uint4*)(base);
    const uint4* p1 = (const uint4*)(base + 8);
    uint4 a = *p0, b = *p1;
    v8u u = { a.x, a.y, a.z, a.w, b.x, b.y, b.z, b.w };
    return __builtin_bit_cast(v16h, u);
}

// B-matrix fragment (32x16 f16). Contiguous k order (same as A): lane L holds column n=L&15;
// lanes 0-15 hold K=0..15, lanes 16-31 hold K=16..31 (contiguous per lane).
// => one contiguous 16-half (32B) run.
__device__ __forceinline__ v16h load_b16(const _Float16* base) {
    const uint4* p = (const uint4*)(base);
    uint4 a = p[0], b = p[1];
    v8u u = { a.x, a.y, a.z, a.w, b.x, b.y, b.z, b.w };
    return __builtin_bit_cast(v16h, u);
}

// ---------------------------------------------------------------------------
// Kernel 0: fp32 -> fp16 weight conversion (one-time, read weights once).
// ---------------------------------------------------------------------------
__global__ __launch_bounds__(256) void convert_weights(
    const float* __restrict__ wq, const float* __restrict__ wk,
    const float* __restrict__ wv, const float* __restrict__ wo,
    _Float16* __restrict__ wqh, _Float16* __restrict__ wkh,
    _Float16* __restrict__ wvh, _Float16* __restrict__ woh)
{
    int i = blockIdx.x * 256 + threadIdx.x;   // 512*512 = 262144 threads
    wqh[i] = (_Float16)wq[i];
    wkh[i] = (_Float16)wk[i];
    wvh[i] = (_Float16)wv[i];
    woh[i] = (_Float16)wo[i];
}

// ---------------------------------------------------------------------------
// Kernel 1: Q/K/V projections.  One block = one (batch, 16-token tile).
// Q,K stored token-major [B,N,C] f16; V stored transposed [B,C,N] f16 so the
// P·V B-matrix gathers in kernel 2 are contiguous.
// ---------------------------------------------------------------------------
__global__ __launch_bounds__(256) void qkv_kernel(
    const float* __restrict__ x,
    const _Float16* __restrict__ wqh, const float* __restrict__ bq,
    const _Float16* __restrict__ wkh, const float* __restrict__ bk,
    const _Float16* __restrict__ wvh, const float* __restrict__ bv,
    _Float16* __restrict__ Q, _Float16* __restrict__ K, _Float16* __restrict__ Vt)
{
    __shared__ _Float16 xt[16 * LROW];   // token-major x tile, padded rows

    const int blk = blockIdx.x;
    const int b   = blk >> 6;
    const int n0  = (blk & 63) << 4;
    const int tid = threadIdx.x;

    // Transpose-load x tile: x[b, c, n0+tn] -> xt[tn][c]   (coalesced over tn)
    for (int idx = tid; idx < 16 * CDIM; idx += 256) {
        int tn = idx & 15, c = idx >> 4;
        xt[tn * LROW + c] = (_Float16)x[(b * CDIM + c) * NTOK + n0 + tn];
    }
    __syncthreads();

    const int lane = tid & 31;
    const int w    = tid >> 5;
    const int col  = lane & 15;     // D column / A row / B column
    const int half = lane >> 4;

    for (int j = 0; j < 12; ++j) {
        const int gj  = w * 12 + j;        // 96 jobs: 3 matrices x 32 o-chunks
        const int mat = gj >> 5;
        const int oc  = gj & 31;
        const _Float16* wsel = (mat == 0) ? wqh : (mat == 1) ? wkh : wvh;
        const float*    bsel = (mat == 0) ? bq  : (mat == 1) ? bk  : bv;

        v8f acc = {0.f, 0.f, 0.f, 0.f, 0.f, 0.f, 0.f, 0.f};
        for (int ks = 0; ks < 16; ++ks) {
            const int c0 = ks * 32;
            v16h a  = load_a16(xt + col * LROW + c0 + half * 16);
            v16h bm = load_b16(wsel + (oc * 16 + col) * CDIM + c0 + half * 16);
            WMMA_F16(acc, a, bm);
        }
        const float bias = bsel[oc * 16 + col];
        #pragma unroll
        for (int r = 0; r < 8; ++r) {
            const int   mrow = r + half * 8;
            const float val  = acc[r] + bias;
            if (mat == 0)
                Q[(b * NTOK + n0 + mrow) * CDIM + oc * 16 + col] = (_Float16)val;
            else if (mat == 1)
                K[(b * NTOK + n0 + mrow) * CDIM + oc * 16 + col] = (_Float16)val;
            else
                Vt[(b * CDIM + oc * 16 + col) * NTOK + n0 + mrow] = (_Float16)val;
        }
    }
}

// ---------------------------------------------------------------------------
// Kernel 2: fused attention + output projection + residual.
// One block = one (batch, 16-query tile).  Dynamic LDS partition:
//   qsv : 16 x LROW f16  (Q tile, later reused for SV tile)
//   sp  : 16 x PROW f16  (scores, then probabilities, in place)
//   red : 16x16 f32 scratch + 16 f32 row stats
// ---------------------------------------------------------------------------
__global__ __launch_bounds__(256) void attn_kernel(
    const float* __restrict__ x,
    const _Float16* __restrict__ Q, const _Float16* __restrict__ K,
    const _Float16* __restrict__ Vt,
    const _Float16* __restrict__ woh, const float* __restrict__ bo,
    float* __restrict__ out)
{
    extern __shared__ char smem_raw[];
    _Float16* qsv = (_Float16*)smem_raw;          // 16*LROW halfs
    _Float16* sp  = qsv + 16 * LROW;              // 16*PROW halfs
    float*    red = (float*)(sp + 16 * PROW);     // 256 floats
    float*    rowm = red + 256;                   // 16 floats

    const int blk = blockIdx.x;
    const int b   = blk >> 6;
    const int n0  = (blk & 63) << 4;
    const int tid = threadIdx.x;
    const int lane = tid & 31;
    const int w    = tid >> 5;
    const int col  = lane & 15;
    const int half = lane >> 4;

    // ---- Async-stage Q tile into LDS: 16 rows x 1024 B, 64 B per thread ----
    // Q rows are contiguous f16; LDS rows padded to LROW halfs (pad lies past
    // the 1024-B payload, so 64-B segments never cross it).  Tracked with
    // ASYNCcnt; drained with s_wait_asynccnt before the block barrier.
    {
        const int row = tid >> 4;      // 0..15
        const int seg = tid & 15;      // 0..15 (64-B segments)
        const _Float16* gsrc = Q + (b * NTOK + n0 + row) * CDIM + seg * 32;
        _Float16* ldst = qsv + row * LROW + seg * 32;
        unsigned lds_off = (unsigned)(unsigned long long)(void*)ldst;
        unsigned long long gaddr = (unsigned long long)gsrc;
        asm volatile(
            "global_load_async_to_lds_b128 %0, %1, off\n\t"
            "global_load_async_to_lds_b128 %0, %1, off offset:16\n\t"
            "global_load_async_to_lds_b128 %0, %1, off offset:32\n\t"
            "global_load_async_to_lds_b128 %0, %1, off offset:48"
            :: "v"(lds_off), "v"(gaddr) : "memory");
        asm volatile("s_wait_asynccnt 0x0" ::: "memory");
    }
    __syncthreads();

    // ---- Pass 1: S = Q K^T  (each wave: 8 key tiles) ----
    for (int kt = w * 8; kt < w * 8 + 8; ++kt) {
        v8f acc = {0.f, 0.f, 0.f, 0.f, 0.f, 0.f, 0.f, 0.f};
        for (int ks = 0; ks < 16; ++ks) {
            const int c0 = ks * 32;
            v16h a  = load_a16(qsv + col * LROW + c0 + half * 16);
            v16h bm = load_b16(K + (b * NTOK + kt * 16 + col) * CDIM + c0 + half * 16);
            WMMA_F16(acc, a, bm);
        }
        #pragma unroll
        for (int r = 0; r < 8; ++r)
            sp[(r + half * 8) * PROW + kt * 16 + col] = (_Float16)acc[r];
    }
    __syncthreads();

    // ---- Softmax over 1024 cols, 16 threads per row, in-place in sp ----
    {
        const int row = tid >> 4, sub = tid & 15;
        const float scale = 0.04419417382415922f;   // 1/sqrt(512)
        float lmax = -1e30f;
        for (int k = 0; k < 64; ++k) {
            float s = (float)sp[row * PROW + sub + k * 16];
            lmax = fmaxf(lmax, s);
        }
        red[row * 16 + sub] = lmax;
        __syncthreads();
        if (sub == 0) {
            float m = red[row * 16];
            for (int i = 1; i < 16; ++i) m = fmaxf(m, red[row * 16 + i]);
            rowm[row] = m;
        }
        __syncthreads();
        const float M = rowm[row];
        float lsum = 0.f;
        for (int k = 0; k < 64; ++k) {
            int idx = row * PROW + sub + k * 16;
            float e = __expf(((float)sp[idx] - M) * scale);
            sp[idx] = (_Float16)e;
            lsum += e;
        }
        __syncthreads();
        red[row * 16 + sub] = lsum;
        __syncthreads();
        if (sub == 0) {
            float s = 0.f;
            for (int i = 0; i < 16; ++i) s += red[row * 16 + i];
            rowm[row] = s;
        }
        __syncthreads();
        const float rinv = 1.0f / rowm[row];
        for (int k = 0; k < 64; ++k) {
            int idx = row * PROW + sub + k * 16;
            sp[idx] = (_Float16)((float)sp[idx] * rinv);
        }
    }
    __syncthreads();

    // ---- Pass 2: SV = P V  (each wave: 4 c-chunks, A shared per K-step) ----
    {
        const int cc0 = w * 4;
        v8f acc0 = {0,0,0,0,0,0,0,0}, acc1 = {0,0,0,0,0,0,0,0};
        v8f acc2 = {0,0,0,0,0,0,0,0}, acc3 = {0,0,0,0,0,0,0,0};
        for (int ks = 0; ks < 32; ++ks) {
            const int t0 = ks * 32;
            v16h a = load_a16(sp + col * PROW + t0 + half * 16);
            v16h b0 = load_b16(Vt + (b * CDIM + (cc0 + 0) * 16 + col) * NTOK + t0 + half * 16);
            v16h b1 = load_b16(Vt + (b * CDIM + (cc0 + 1) * 16 + col) * NTOK + t0 + half * 16);
            v16h b2 = load_b16(Vt + (b * CDIM + (cc0 + 2) * 16 + col) * NTOK + t0 + half * 16);
            v16h b3 = load_b16(Vt + (b * CDIM + (cc0 + 3) * 16 + col) * NTOK + t0 + half * 16);
            WMMA_F16(acc0, a, b0);
            WMMA_F16(acc1, a, b1);
            WMMA_F16(acc2, a, b2);
            WMMA_F16(acc3, a, b3);
        }
        // Stage SV tile into qsv (Q tile no longer needed) as f16
        #pragma unroll
        for (int r = 0; r < 8; ++r) {
            const int mrow = r + half * 8;
            qsv[mrow * LROW + (cc0 + 0) * 16 + col] = (_Float16)acc0[r];
            qsv[mrow * LROW + (cc0 + 1) * 16 + col] = (_Float16)acc1[r];
            qsv[mrow * LROW + (cc0 + 2) * 16 + col] = (_Float16)acc2[r];
            qsv[mrow * LROW + (cc0 + 3) * 16 + col] = (_Float16)acc3[r];
        }
    }
    __syncthreads();

    // ---- Pass 3: out = SV Wo^T + bo + x  (each wave: 4 o-chunks) ----
    for (int q2 = 0; q2 < 4; ++q2) {
        const int oc = w * 4 + q2;
        v8f acc = {0.f, 0.f, 0.f, 0.f, 0.f, 0.f, 0.f, 0.f};
        for (int ks = 0; ks < 16; ++ks) {
            const int c0 = ks * 32;
            v16h a  = load_a16(qsv + col * LROW + c0 + half * 16);
            v16h bm = load_b16(woh + (oc * 16 + col) * CDIM + c0 + half * 16);
            WMMA_F16(acc, a, bm);
        }
        const float bias = bo[oc * 16 + col];
        #pragma unroll
        for (int r = 0; r < 8; ++r) {
            const int mrow = r + half * 8;
            const int gi = (b * CDIM + oc * 16 + col) * NTOK + n0 + mrow;
            out[gi] = x[gi] + acc[r] + bias;
        }
    }
}

// ---------------------------------------------------------------------------
// Host launcher
// ---------------------------------------------------------------------------
extern "C" void kernel_launch(void* const* d_in, const int* in_sizes, int n_in,
                              void* d_out, int out_size, void* d_ws, size_t ws_size,
                              hipStream_t stream) {
    const float* x  = (const float*)d_in[0];
    const float* wq = (const float*)d_in[1];
    const float* bq = (const float*)d_in[2];
    const float* wk = (const float*)d_in[3];
    const float* bk = (const float*)d_in[4];
    const float* wv = (const float*)d_in[5];
    const float* bv = (const float*)d_in[6];
    const float* wo = (const float*)d_in[7];
    const float* bo = (const float*)d_in[8];
    float* out = (float*)d_out;

    char* ws = (char*)d_ws;
    const size_t WBYTES = (size_t)CDIM * CDIM * sizeof(_Float16);        // 512 KB
    const size_t TBYTES = (size_t)BATCH * NTOK * CDIM * sizeof(_Float16); // 16 MB
    _Float16* wqh = (_Float16*)(ws);
    _Float16* wkh = (_Float16*)(ws + WBYTES);
    _Float16* wvh = (_Float16*)(ws + 2 * WBYTES);
    _Float16* woh = (_Float16*)(ws + 3 * WBYTES);
    _Float16* Qh  = (_Float16*)(ws + 4 * WBYTES);
    _Float16* Kh  = (_Float16*)(ws + 4 * WBYTES + TBYTES);
    _Float16* Vth = (_Float16*)(ws + 4 * WBYTES + 2 * TBYTES);

    convert_weights<<<dim3((CDIM * CDIM) / 256), dim3(256), 0, stream>>>(
        wq, wk, wv, wo, wqh, wkh, wvh, woh);

    qkv_kernel<<<dim3(BATCH * (NTOK / 16)), dim3(256), 0, stream>>>(
        x, wqh, bq, wkh, bk, wvh, bv, Qh, Kh, Vth);

    const size_t smem2 = (size_t)(16 * LROW + 16 * PROW) * sizeof(_Float16)
                       + (size_t)(256 + 16) * sizeof(float);
    attn_kernel<<<dim3(BATCH * (NTOK / 16)), dim3(256), smem2, stream>>>(
        x, Qh, Kh, Vth, woh, bo, out);
}
